// CombinatorialNER_69183333204262
// MI455X (gfx1250) — matrix operand order repaired and checked by reading, not compiled, not run on hardware
//
#include <hip/hip_runtime.h>

// ---------------------------------------------------------------------------
// CDNA5 (gfx1250) bf16 WMMA helpers (wave32)
// ---------------------------------------------------------------------------
typedef __attribute__((ext_vector_type(16))) __bf16 v16bf;
typedef __attribute__((ext_vector_type(8)))  float  v8f;

union NerFrag16 { v16bf v; unsigned int u[8]; };

__device__ __forceinline__ unsigned short ner_f2bf(float f) {
    unsigned int u = __float_as_uint(f);
    unsigned int r = (u + 0x7FFFu + ((u >> 16) & 1u)) >> 16;   // RNE
    return (unsigned short)r;
}

// 16-bit A-matrix 16x32 K offset per VGPR j (pairs contiguous in K)
__device__ __forceinline__ int ner_koffA(int j, int half) {
    return ((j >> 2) << 4) + (half << 3) + ((j & 3) << 1);
}
// 16-bit B-matrix 32x16 K offset per VGPR j
__device__ __forceinline__ int ner_koffB(int j, int half) {
    return ((j >> 2) << 4) + (half << 3) + ((j & 3) << 1);
}

// A fragment: rows m = lane&15 from row-major [M][ld] bf16, tile base p = &A[row0][k0]
__device__ __forceinline__ v16bf ner_load_A(const unsigned short* p, int ld) {
    int lane = threadIdx.x & 31, half = lane >> 4, m = lane & 15;
    NerFrag16 f;
    const unsigned short* row = p + (size_t)m * ld;
#pragma unroll
    for (int j = 0; j < 8; ++j)
        f.u[j] = *(const unsigned int*)(row + ner_koffA(j, half));
    return f.v;
}

// B fragment: weights stored row-major (N,K) -> B[k][n] = W[n0+n][k0+k]
__device__ __forceinline__ v16bf ner_load_B(const unsigned short* p, int ld) {
    int lane = threadIdx.x & 31, half = lane >> 4, n = lane & 15;
    NerFrag16 f;
    const unsigned short* row = p + (size_t)n * ld;
#pragma unroll
    for (int j = 0; j < 8; ++j)
        f.u[j] = *(const unsigned int*)(row + ner_koffB(j, half));
    return f.v;
}

__device__ __forceinline__ v8f ner_wmma(v16bf a, v16bf b, v8f c) {
    return __builtin_amdgcn_wmma_f32_16x16x32_bf16(false, a, false, b, (short)0, c, false, false);
}

__device__ __forceinline__ float ner_sig(float x) { return 1.0f / (1.0f + __expf(-x)); }

// ---------------------------------------------------------------------------
// K1: pad + convert a (N,K) f32 weight to (Npad,Kpad) bf16, zero-filled
// ---------------------------------------------------------------------------
__global__ void ner_prep_weight(const float* __restrict__ src, unsigned short* __restrict__ dst,
                                int N, int K, int Npad, int Kpad) {
    int total = Npad * Kpad;
    for (int i = blockIdx.x * blockDim.x + threadIdx.x; i < total; i += gridDim.x * blockDim.x) {
        int n = i / Kpad, k = i % Kpad;
        float v = (n < N && k < K) ? src[(size_t)n * K + k] : 0.0f;
        dst[i] = ner_f2bf(v);
    }
}

// ---------------------------------------------------------------------------
// K2: per-token word-embedding gather + char CNN feats + char-LSTM emb gather
//     combined[tok] layout (bf16, 416 cols): [we 0..199 | cnn 200..295 | lstm 296..395 | pad]
// ---------------------------------------------------------------------------
__global__ void ner_embed_cnn(const int* __restrict__ word_ids, const int* __restrict__ char_ids,
                              const float* __restrict__ word_emb, const float* __restrict__ char_emb_cnn,
                              const float* __restrict__ char_emb_lstm,
                              const float* __restrict__ cw3, const float* __restrict__ cb3,
                              const float* __restrict__ cw5, const float* __restrict__ cb5,
                              const float* __restrict__ cw7, const float* __restrict__ cb7,
                              unsigned short* __restrict__ combined, unsigned short* __restrict__ cel) {
    int tok = blockIdx.x, tid = threadIdx.x;            // 128 threads
    __shared__ float ce[30][16];
    __shared__ int cid[16];
    if (tid < 16) cid[tid] = char_ids[(size_t)tok * 16 + tid];
    __syncthreads();
    for (int i = tid; i < 30 * 16; i += 128) {
        int ch = i >> 4, w = i & 15;
        ce[ch][w] = char_emb_cnn[(size_t)cid[w] * 30 + ch];
    }
    for (int i = tid; i < 16 * 32; i += 128) {          // cel bf16, K padded 30->32
        int w = i >> 5, k = i & 31;
        float v = (k < 30) ? char_emb_lstm[(size_t)cid[w] * 30 + k] : 0.0f;
        cel[(size_t)tok * 512 + i] = ner_f2bf(v);
    }
    int wid = word_ids[tok];
    for (int i = tid; i < 200; i += 128)
        combined[(size_t)tok * 416 + i] = ner_f2bf(word_emb[(size_t)wid * 200 + i]);
    for (int i = 396 + tid; i < 416; i += 128)
        combined[(size_t)tok * 416 + i] = 0;            // zero K padding
    __syncthreads();
    if (tid < 96) {                                     // (conv, out-channel)
        int ci = tid >> 5, oc = tid & 31;
        int ksz = 3 + 2 * ci, hw = ksz >> 1;
        const float* w  = (ci == 0) ? cw3 : (ci == 1) ? cw5 : cw7;
        const float* bb = (ci == 0) ? cb3 : (ci == 1) ? cb5 : cb7;
        float bias = bb[oc], mx = 0.0f;                 // relu(...)>=0 so 0 init ok
        for (int pos = 0; pos < 16; ++pos) {
            float acc = bias;
            for (int tap = 0; tap < ksz; ++tap) {
                int p = pos + tap - hw;
                if (p < 0 || p >= 16) continue;
                for (int ic = 0; ic < 30; ++ic)
                    acc += w[((size_t)oc * 30 + ic) * ksz + tap] * ce[ic][p];
            }
            float rel = acc > 0.0f ? acc : 0.0f;
            mx = rel > mx ? rel : mx;
        }
        combined[(size_t)tok * 416 + 200 + ci * 32 + oc] = ner_f2bf(mx);
    }
}

// ---------------------------------------------------------------------------
// K3: char BiLSTM. One wave per 16-token tile, 16 sequential steps.
//     gates (16,208) = WMMA( [x|h] (K=32+64) , W (208,K) ).  grid=(1024,2dir)
// ---------------------------------------------------------------------------
__global__ void ner_char_lstm(const unsigned short* __restrict__ cel,
                              const unsigned short* __restrict__ wih_f, const unsigned short* __restrict__ whh_f,
                              const float* __restrict__ bih_f, const float* __restrict__ bhh_f,
                              const unsigned short* __restrict__ wih_b, const unsigned short* __restrict__ whh_b,
                              const float* __restrict__ bih_b, const float* __restrict__ bhh_b,
                              unsigned short* __restrict__ combined) {
    int dir = blockIdx.y;
    int r0 = blockIdx.x * 16;
    const unsigned short* wih = dir ? wih_b : wih_f;
    const unsigned short* whh = dir ? whh_b : whh_f;
    const float* bih = dir ? bih_b : bih_f;
    const float* bhh = dir ? bhh_b : bhh_f;
    __shared__ unsigned short h_bf[16][64];             // hidden, K padded 50->64
    __shared__ float c_st[16][50];
    __shared__ float gates[16][208];
    int lane = threadIdx.x;                             // 32 threads (1 wave)
    for (int i = lane; i < 16 * 64; i += 32) h_bf[i >> 6][i & 63] = 0;
    for (int i = lane; i < 16 * 50; i += 32) c_st[i / 50][i % 50] = 0.0f;
    __syncthreads();
    int half = lane >> 4, nn = lane & 15;
    for (int s = 0; s < 16; ++s) {
        int w = dir ? (15 - s) : s;
        v16bf ax  = ner_load_A(cel + (size_t)r0 * 512 + w * 32, 512);
        v16bf ah0 = ner_load_A(&h_bf[0][0], 64);
        v16bf ah1 = ner_load_A(&h_bf[0][0] + 32, 64);
        for (int nt = 0; nt < 13; ++nt) {
            int n0 = nt * 16;
            v8f acc = {0.f,0.f,0.f,0.f,0.f,0.f,0.f,0.f};
            acc = ner_wmma(ax,  ner_load_B(wih + (size_t)n0 * 32, 32), acc);
            acc = ner_wmma(ah0, ner_load_B(whh + (size_t)n0 * 64, 64), acc);
            acc = ner_wmma(ah1, ner_load_B(whh + (size_t)n0 * 64 + 32, 64), acc);
#pragma unroll
            for (int r = 0; r < 8; ++r) gates[half * 8 + r][n0 + nn] = acc[r];
        }
        __syncthreads();
        for (int idx = lane; idx < 16 * 50; idx += 32) {
            int m = idx / 50, j = idx % 50;
            float gi = gates[m][j]        + bih[j]        + bhh[j];
            float gf = gates[m][50 + j]   + bih[50 + j]   + bhh[50 + j];
            float gg = gates[m][100 + j]  + bih[100 + j]  + bhh[100 + j];
            float go = gates[m][150 + j]  + bih[150 + j]  + bhh[150 + j];
            float cc = ner_sig(gf) * c_st[m][j] + ner_sig(gi) * tanhf(gg);
            c_st[m][j] = cc;
            h_bf[m][j] = ner_f2bf(ner_sig(go) * tanhf(cc));
        }
        __syncthreads();
    }
    int base = 296 + dir * 50;                          // hf -> 296.., hb -> 346..
    for (int idx = lane; idx < 16 * 50; idx += 32) {
        int m = idx / 50, j = idx % 50;
        combined[(size_t)(r0 + m) * 416 + base + j] = h_bf[m][j];
    }
}

// ---------------------------------------------------------------------------
// K4: fuse layer GEMM: (16384,416)bf16 @ (208,416)bf16 -> relu -> fused (16384,224)bf16
// ---------------------------------------------------------------------------
__global__ void ner_fuse_gemm(const unsigned short* __restrict__ combined,
                              const unsigned short* __restrict__ w_fuse,
                              const float* __restrict__ fuse_b, unsigned short* __restrict__ fused) {
    int wave = threadIdx.x >> 5, lane = threadIdx.x & 31;
    int m0 = blockIdx.x * 64 + wave * 16;
    int half = lane >> 4, nn = lane & 15;
    for (int nt = 0; nt < 13; ++nt) {
        int n0 = nt * 16;
        v8f acc = {0.f,0.f,0.f,0.f,0.f,0.f,0.f,0.f};
        for (int kt = 0; kt < 13; ++kt) {
            v16bf a = ner_load_A(combined + (size_t)m0 * 416 + kt * 32, 416);
            v16bf b = ner_load_B(w_fuse + (size_t)n0 * 416 + kt * 32, 416);
            acc = ner_wmma(a, b, acc);
        }
        int col = n0 + nn;
        if (col < 200) {
            float bb = fuse_b[col];
#pragma unroll
            for (int r = 0; r < 8; ++r) {
                float v = acc[r] + bb;
                v = v > 0.0f ? v : 0.0f;
                fused[(size_t)(m0 + half * 8 + r) * 224 + col] = ner_f2bf(v);
            }
        }
    }
    for (int i = lane; i < 16 * 24; i += 32) {          // zero K padding 200..223
        int m = i / 24, col = 200 + i % 24;
        fused[(size_t)(m0 + m) * 224 + col] = 0;
    }
}

// ---------------------------------------------------------------------------
// K5: ctx-LSTM input projection GEMM (hoisted out of the recurrence):
//     xproj[dir] = fused @ wih.T + (bih+bhh), (16384,512) f32.  grid=(256,2)
// ---------------------------------------------------------------------------
__global__ void ner_ctx_xproj(const unsigned short* __restrict__ fused,
                              const unsigned short* __restrict__ wih_f,
                              const float* __restrict__ bih_f, const float* __restrict__ bhh_f,
                              const unsigned short* __restrict__ wih_b,
                              const float* __restrict__ bih_b, const float* __restrict__ bhh_b,
                              float* __restrict__ xproj_f, float* __restrict__ xproj_b) {
    int dir = blockIdx.y;
    const unsigned short* wih = dir ? wih_b : wih_f;
    const float* bih = dir ? bih_b : bih_f;
    const float* bhh = dir ? bhh_b : bhh_f;
    float* xp = dir ? xproj_b : xproj_f;
    int wave = threadIdx.x >> 5, lane = threadIdx.x & 31;
    int m0 = blockIdx.x * 64 + wave * 16;
    int half = lane >> 4, nn = lane & 15;
    for (int nt = 0; nt < 32; ++nt) {
        int n0 = nt * 16;
        v8f acc = {0.f,0.f,0.f,0.f,0.f,0.f,0.f,0.f};
        for (int kt = 0; kt < 7; ++kt) {
            v16bf a = ner_load_A(fused + (size_t)m0 * 224 + kt * 32, 224);
            v16bf b = ner_load_B(wih + (size_t)n0 * 224 + kt * 32, 224);
            acc = ner_wmma(a, b, acc);
        }
        int col = n0 + nn;
        float bb = bih[col] + bhh[col];
#pragma unroll
        for (int r = 0; r < 8; ++r)
            xp[(size_t)(m0 + half * 8 + r) * 512 + col] = acc[r] + bb;
    }
}

// ---------------------------------------------------------------------------
// K6: ctx BiLSTM recurrence. One persistent workgroup per direction (16 waves),
//     h/c/gates LDS-resident; per step: gates = xproj[t] + h @ whh.T via WMMA.
// ---------------------------------------------------------------------------
__global__ void __launch_bounds__(512) ner_ctx_recur(
        const float* __restrict__ xproj_f, const float* __restrict__ xproj_b,
        const unsigned short* __restrict__ whh_f, const unsigned short* __restrict__ whh_b,
        float* __restrict__ H, unsigned short* __restrict__ Hbf) {
    int dir = blockIdx.x;
    const float* xp = dir ? xproj_b : xproj_f;
    const unsigned short* whh = dir ? whh_b : whh_f;
    extern __shared__ char smem[];
    float* gates = (float*)smem;                                   // 64*512 f32
    float* c_st  = gates + 64 * 512;                               // 64*128 f32
    unsigned short* h_bf = (unsigned short*)(c_st + 64 * 128);     // 64*128 bf16
    int tid = threadIdx.x;
    for (int i = tid; i < 64 * 128; i += 512) { c_st[i] = 0.0f; h_bf[i] = 0; }
    __syncthreads();
    int wave = tid >> 5, lane = tid & 31, half = lane >> 4, nn = lane & 15;
    for (int s = 0; s < 256; ++s) {
        int t = dir ? (255 - s) : s;
        for (int q = 0; q < 8; ++q) {                              // 8 of 128 tiles per wave
            int tile = wave * 8 + q;
            int m0 = (tile >> 5) * 16, n0 = (tile & 31) * 16;
            v8f acc = {0.f,0.f,0.f,0.f,0.f,0.f,0.f,0.f};
            for (int kt = 0; kt < 4; ++kt) {
                v16bf a = ner_load_A(h_bf + (size_t)m0 * 128 + kt * 32, 128);
                v16bf b = ner_load_B(whh + (size_t)n0 * 128 + kt * 32, 128);
                acc = ner_wmma(a, b, acc);
            }
            int col = n0 + nn;
#pragma unroll
            for (int r = 0; r < 8; ++r) {
                int brow = m0 + half * 8 + r;
                gates[(size_t)brow * 512 + col] =
                    acc[r] + xp[((size_t)brow * 256 + t) * 512 + col];
            }
        }
        __syncthreads();
        for (int i = tid; i < 64 * 128; i += 512) {
            int b = i >> 7, j = i & 127;
            float gi = gates[(size_t)b * 512 + j];
            float gf = gates[(size_t)b * 512 + 128 + j];
            float gg = gates[(size_t)b * 512 + 256 + j];
            float go = gates[(size_t)b * 512 + 384 + j];
            float cc = ner_sig(gf) * c_st[i] + ner_sig(gi) * tanhf(gg);
            c_st[i] = cc;
            float hh = ner_sig(go) * tanhf(cc);
            h_bf[i] = ner_f2bf(hh);
            size_t orow = ((size_t)b * 256 + t) * 256 + (size_t)dir * 128 + j;
            H[orow] = hh;
            Hbf[orow] = ner_f2bf(hh);
        }
        __syncthreads();
    }
}

// ---------------------------------------------------------------------------
// K7: attention scores: sum_n tanh(H @ attn_w.T + b)[n] * v[n] per token (WMMA)
// ---------------------------------------------------------------------------
__global__ void ner_attn_score(const unsigned short* __restrict__ Hbf,
                               const unsigned short* __restrict__ w_attn,
                               const float* __restrict__ attn_b, const float* __restrict__ attn_v,
                               float* __restrict__ scores) {
    int wave = threadIdx.x >> 5, lane = threadIdx.x & 31;
    int m0 = blockIdx.x * 64 + wave * 16;
    int half = lane >> 4, nn = lane & 15;
    float rs[8];
#pragma unroll
    for (int r = 0; r < 8; ++r) rs[r] = 0.0f;
    for (int nt = 0; nt < 16; ++nt) {
        int n0 = nt * 16;
        v8f acc = {0.f,0.f,0.f,0.f,0.f,0.f,0.f,0.f};
        for (int kt = 0; kt < 8; ++kt) {
            v16bf a = ner_load_A(Hbf + (size_t)m0 * 256 + kt * 32, 256);
            v16bf b = ner_load_B(w_attn + (size_t)n0 * 256 + kt * 32, 256);
            acc = ner_wmma(a, b, acc);
        }
        int col = n0 + nn;
        float bb = attn_b[col], vv = attn_v[col];
#pragma unroll
        for (int r = 0; r < 8; ++r) rs[r] += tanhf(acc[r] + bb) * vv;
    }
#pragma unroll
    for (int r = 0; r < 8; ++r) {                       // reduce 16 lanes per half-wave
        float v = rs[r];
        for (int off = 1; off < 16; off <<= 1) v += __shfl_xor(v, off, 32);
        if (nn == 0) scores[m0 + half * 8 + r] = v;
    }
}

// ---------------------------------------------------------------------------
// K8: masked softmax over T per batch row, scale H, emission scores (K=9, VALU)
// ---------------------------------------------------------------------------
__global__ void ner_softmax_emis(const float* __restrict__ H, const float* __restrict__ scores,
                                 const unsigned char* __restrict__ mask,
                                 const float* __restrict__ emis_w, const float* __restrict__ emis_b,
                                 float* __restrict__ emis) {
    int b = blockIdx.x, t = threadIdx.x;                // 256 threads
    __shared__ float red[256];
    int tok = b * 256 + t;
    float s = mask[tok] ? scores[tok] : -1.0e9f;
    red[t] = s; __syncthreads();
    for (int off = 128; off > 0; off >>= 1) {
        if (t < off) red[t] = fmaxf(red[t], red[t + off]);
        __syncthreads();
    }
    float mx = red[0]; __syncthreads();
    float e = __expf(s - mx);
    red[t] = e; __syncthreads();
    for (int off = 128; off > 0; off >>= 1) {
        if (t < off) red[t] += red[t + off];
        __syncthreads();
    }
    float p = e / red[0];
    const float* hrow = H + (size_t)tok * 256;
    for (int k = 0; k < 9; ++k) {
        float acc = emis_b[k];
        const float* wr = emis_w + (size_t)k * 256;
        for (int n = 0; n < 256; ++n) acc += hrow[n] * p * wr[n];
        emis[(size_t)tok * 9 + k] = acc;
    }
}

// ---------------------------------------------------------------------------
// K9: CRF numerator + forward algorithm (logsumexp over 9 states), mean loss
// ---------------------------------------------------------------------------
__global__ void ner_crf_loss(const float* __restrict__ emis, const unsigned char* __restrict__ mask,
                             const int* __restrict__ tags,
                             const float* __restrict__ crf_start, const float* __restrict__ crf_end,
                             const float* __restrict__ crf_trans, float* __restrict__ out) {
    __shared__ float trans[81];
    __shared__ float losses[64];
    int tid = threadIdx.x;                              // 64 threads
    for (int i = tid; i < 81; i += 64) trans[i] = crf_trans[i];
    __syncthreads();
    int b = tid;
    const float* em = emis + (size_t)b * 256 * 9;
    const int* tg = tags + (size_t)b * 256;
    const unsigned char* mk = mask + (size_t)b * 256;
    int prev = tg[0], last = 0;
    float num = crf_start[prev] + em[prev];
    for (int t = 1; t < 256; ++t) {
        int cur = tg[t];
        if (mk[t]) { num += trans[prev * 9 + cur] + em[(size_t)t * 9 + cur]; last = t; }
        prev = cur;
    }
    num += crf_end[tg[last]];
    float alpha[9];
#pragma unroll
    for (int k = 0; k < 9; ++k) alpha[k] = crf_start[k] + em[k];
    for (int t = 1; t < 256; ++t) {
        if (!mk[t]) continue;
        float nxt[9];
        for (int k2 = 0; k2 < 9; ++k2) {
            float mx = -1.0e30f;
            for (int k1 = 0; k1 < 9; ++k1) mx = fmaxf(mx, alpha[k1] + trans[k1 * 9 + k2]);
            float ss = 0.0f;
            for (int k1 = 0; k1 < 9; ++k1) ss += __expf(alpha[k1] + trans[k1 * 9 + k2] - mx);
            nxt[k2] = mx + __logf(ss) + em[(size_t)t * 9 + k2];
        }
#pragma unroll
        for (int k = 0; k < 9; ++k) alpha[k] = nxt[k];
    }
    float mx = -1.0e30f;
    for (int k = 0; k < 9; ++k) mx = fmaxf(mx, alpha[k] + crf_end[k]);
    float ss = 0.0f;
    for (int k = 0; k < 9; ++k) ss += __expf(alpha[k] + crf_end[k] - mx);
    losses[b] = num - (mx + __logf(ss));
    __syncthreads();
    if (tid == 0) {
        float acc = 0.0f;
        for (int i = 0; i < 64; ++i) acc += losses[i];
        out[0] = -(acc / 64.0f);
    }
}

// ---------------------------------------------------------------------------
// Launch
// ---------------------------------------------------------------------------
extern "C" void kernel_launch(void* const* d_in, const int* in_sizes, int n_in,
                              void* d_out, int out_size, void* d_ws, size_t ws_size,
                              hipStream_t stream) {
    (void)in_sizes; (void)n_in; (void)out_size; (void)ws_size;
    const int*   word_ids      = (const int*)  d_in[0];
    const int*   char_ids      = (const int*)  d_in[1];
    const unsigned char* mask  = (const unsigned char*)d_in[2];
    const int*   tags          = (const int*)  d_in[3];
    const float* word_emb      = (const float*)d_in[4];
    const float* char_emb_cnn  = (const float*)d_in[5];
    const float* conv_w3 = (const float*)d_in[6],  *conv_b3 = (const float*)d_in[7];
    const float* conv_w5 = (const float*)d_in[8],  *conv_b5 = (const float*)d_in[9];
    const float* conv_w7 = (const float*)d_in[10], *conv_b7 = (const float*)d_in[11];
    const float* char_emb_lstm = (const float*)d_in[12];
    const float* cl_wih_f = (const float*)d_in[13], *cl_whh_f = (const float*)d_in[14];
    const float* cl_bih_f = (const float*)d_in[15], *cl_bhh_f = (const float*)d_in[16];
    const float* cl_wih_b = (const float*)d_in[17], *cl_whh_b = (const float*)d_in[18];
    const float* cl_bih_b = (const float*)d_in[19], *cl_bhh_b = (const float*)d_in[20];
    const float* fuse_w = (const float*)d_in[21], *fuse_b = (const float*)d_in[22];
    const float* ctx_wih_f = (const float*)d_in[23], *ctx_whh_f = (const float*)d_in[24];
    const float* ctx_bih_f = (const float*)d_in[25], *ctx_bhh_f = (const float*)d_in[26];
    const float* ctx_wih_b = (const float*)d_in[27], *ctx_whh_b = (const float*)d_in[28];
    const float* ctx_bih_b = (const float*)d_in[29], *ctx_bhh_b = (const float*)d_in[30];
    const float* attn_w = (const float*)d_in[31], *attn_b = (const float*)d_in[32];
    const float* attn_v = (const float*)d_in[33];
    const float* emis_w = (const float*)d_in[34], *emis_b = (const float*)d_in[35];
    const float* crf_start = (const float*)d_in[36], *crf_end = (const float*)d_in[37];
    const float* crf_trans = (const float*)d_in[38];
    float* out = (float*)d_out;

    // ---- workspace partition (256B aligned) ----
    char* ws = (char*)d_ws;
    size_t off = 0;
    auto alloc = [&](size_t bytes) -> void* {
        void* p = ws + off;
        off += (bytes + 255) & ~(size_t)255;
        return p;
    };
    unsigned short* w_cl_wih_f = (unsigned short*)alloc(208 * 32 * 2);
    unsigned short* w_cl_whh_f = (unsigned short*)alloc(208 * 64 * 2);
    unsigned short* w_cl_wih_b = (unsigned short*)alloc(208 * 32 * 2);
    unsigned short* w_cl_whh_b = (unsigned short*)alloc(208 * 64 * 2);
    unsigned short* w_fuse     = (unsigned short*)alloc((size_t)208 * 416 * 2);
    unsigned short* w_ctx_wih_f= (unsigned short*)alloc((size_t)512 * 224 * 2);
    unsigned short* w_ctx_wih_b= (unsigned short*)alloc((size_t)512 * 224 * 2);
    unsigned short* w_ctx_whh_f= (unsigned short*)alloc((size_t)512 * 128 * 2);
    unsigned short* w_ctx_whh_b= (unsigned short*)alloc((size_t)512 * 128 * 2);
    unsigned short* w_attn     = (unsigned short*)alloc((size_t)256 * 256 * 2);
    unsigned short* cel        = (unsigned short*)alloc((size_t)16384 * 512 * 2);
    unsigned short* combined   = (unsigned short*)alloc((size_t)16384 * 416 * 2);
    unsigned short* fused      = (unsigned short*)alloc((size_t)16384 * 224 * 2);
    float* xproj_f             = (float*)alloc((size_t)16384 * 512 * 4);
    float* xproj_b             = (float*)alloc((size_t)16384 * 512 * 4);
    float* Hbuf                = (float*)alloc((size_t)16384 * 256 * 4);
    unsigned short* Hbf        = (unsigned short*)alloc((size_t)16384 * 256 * 2);
    float* scores              = (float*)alloc((size_t)16384 * 4);
    float* emis                = (float*)alloc((size_t)16384 * 9 * 4);

    auto prep = [&](const float* s, unsigned short* d, int N, int K, int Np, int Kp) {
        int total = Np * Kp;
        ner_prep_weight<<<(total + 255) / 256, 256, 0, stream>>>(s, d, N, K, Np, Kp);
    };
    prep(cl_wih_f, w_cl_wih_f, 200, 30, 208, 32);
    prep(cl_whh_f, w_cl_whh_f, 200, 50, 208, 64);
    prep(cl_wih_b, w_cl_wih_b, 200, 30, 208, 32);
    prep(cl_whh_b, w_cl_whh_b, 200, 50, 208, 64);
    prep(fuse_w,   w_fuse,     200, 396, 208, 416);
    prep(ctx_wih_f, w_ctx_wih_f, 512, 200, 512, 224);
    prep(ctx_wih_b, w_ctx_wih_b, 512, 200, 512, 224);
    prep(ctx_whh_f, w_ctx_whh_f, 512, 128, 512, 128);
    prep(ctx_whh_b, w_ctx_whh_b, 512, 128, 512, 128);
    prep(attn_w,    w_attn,      256, 256, 256, 256);

    ner_embed_cnn<<<16384, 128, 0, stream>>>(word_ids, char_ids, word_emb, char_emb_cnn,
        char_emb_lstm, conv_w3, conv_b3, conv_w5, conv_b5, conv_w7, conv_b7, combined, cel);

    ner_char_lstm<<<dim3(1024, 2), 32, 0, stream>>>(cel,
        w_cl_wih_f, w_cl_whh_f, cl_bih_f, cl_bhh_f,
        w_cl_wih_b, w_cl_whh_b, cl_bih_b, cl_bhh_b, combined);

    ner_fuse_gemm<<<256, 128, 0, stream>>>(combined, w_fuse, fuse_b, fused);

    ner_ctx_xproj<<<dim3(256, 2), 128, 0, stream>>>(fused,
        w_ctx_wih_f, ctx_bih_f, ctx_bhh_f,
        w_ctx_wih_b, ctx_bih_b, ctx_bhh_b, xproj_f, xproj_b);

    size_t recur_smem = (size_t)64 * 512 * 4 + (size_t)64 * 128 * 4 + (size_t)64 * 128 * 2;
    ner_ctx_recur<<<2, 512, recur_smem, stream>>>(xproj_f, xproj_b,
        w_ctx_whh_f, w_ctx_whh_b, Hbuf, Hbf);

    ner_attn_score<<<256, 128, 0, stream>>>(Hbf, w_attn, attn_b, attn_v, scores);

    ner_softmax_emis<<<64, 256, 0, stream>>>(Hbuf, scores, mask, emis_w, emis_b, emis);

    ner_crf_loss<<<1, 64, 0, stream>>>(emis, mask, tags, crf_start, crf_end, crf_trans, out);
}
